// DeformConv2d_12506944765975
// MI455X (gfx1250) — compile-verified, matrix operand order checked by reading, not run on hardware
//
#include <hip/hip_runtime.h>
#include <hip/hip_bf16.h>

// ---------------- problem constants ----------------
#define NB   4
#define CDIM 256
#define HH   64
#define WW   64
#define LL   (HH * WW)      // 4096
#define MM   (NB * LL)      // 16384
#define GG   4
#define CGRP (CDIM / GG)    // 64
#define PP   9
#define PWO  112
#define PADK 1

typedef _Float16 half8  __attribute__((ext_vector_type(8)));
typedef _Float16 v16h   __attribute__((ext_vector_type(16)));
typedef float    v8f    __attribute__((ext_vector_type(8)));

// -------- helpers --------
__device__ __forceinline__ v16h load_operand16(const _Float16* base) {
    // two contiguous 16B loads -> 8 VGPRs, matching 16-bit WMMA A/B lane layout
    union { v16h v; half8 h[2]; } u;
    u.h[0] = *reinterpret_cast<const half8*>(base);
    u.h[1] = *reinterpret_cast<const half8*>(base + 16);
    return u.v;
}

template <int NT>
__device__ __forceinline__ void load_b_frags(v16h (&b)[NT], const _Float16* bP, int k, int Kd) {
#pragma unroll
    for (int j = 0; j < NT; j++) b[j] = load_operand16(bP + (size_t)j * 16 * Kd + k);
}

__device__ __forceinline__ void load_a_frags(v16h (&a)[4], const _Float16* aP, int k, int Kd) {
#pragma unroll
    for (int i = 0; i < 4; i++) a[i] = load_operand16(aP + (size_t)i * 16 * Kd + k);
}

template <int NT>
__device__ __forceinline__ void mma_step(v8f (&acc)[4][NT], const v16h (&a)[4], const v16h (&b)[NT]) {
#pragma unroll
    for (int i = 0; i < 4; i++)
#pragma unroll
        for (int j = 0; j < NT; j++)
            acc[i][j] = __builtin_amdgcn_wmma_f32_16x16x32_f16(
                false, a[i], false, b[j], (short)0, acc[i][j], false, false);
}

// ---------------- stage kernels ----------------

// x [N,C,L] f32  ->  xT [N,L,C] f16   (coalesced reads over l)
__global__ void k_xpose_f16(const float* __restrict__ x, _Float16* __restrict__ xT) {
    size_t id = (size_t)blockIdx.x * blockDim.x + threadIdx.x;
    if (id >= (size_t)NB * CDIM * LL) return;
    int l = (int)(id % LL);
    int c = (int)((id / LL) % CDIM);
    int n = (int)(id / ((size_t)LL * CDIM));
    float v = x[id];
    xT[((size_t)n * LL + l) * CDIM + c] = (_Float16)v;
}

// flat f32 -> f16 convert (weights)
__global__ void k_cvt_f16(const float* __restrict__ in, _Float16* __restrict__ out, int n) {
    int i = blockIdx.x * blockDim.x + threadIdx.x;
    if (i < n) out[i] = (_Float16)in[i];
}

// depthwise 3x3 conv (pad 1) on x, write transposed f16 [N,L,C]
__global__ void k_dwconv(const float* __restrict__ x, const float* __restrict__ dww,
                         const float* __restrict__ dwb, _Float16* __restrict__ outT) {
    size_t id = (size_t)blockIdx.x * blockDim.x + threadIdx.x;
    if (id >= (size_t)NB * CDIM * LL) return;
    int l = (int)(id % LL);
    int c = (int)((id / LL) % CDIM);
    int n = (int)(id / ((size_t)LL * CDIM));
    int y = l >> 6, xx = l & 63;
    const float* xp = x + ((size_t)n * CDIM + c) * LL;
    const float* wp = dww + c * 9;
    float a = dwb[c];
#pragma unroll
    for (int ky = 0; ky < 3; ky++) {
        int iy = y + ky - PADK;
        if (iy < 0 || iy >= HH) continue;
#pragma unroll
        for (int kx = 0; kx < 3; kx++) {
            int ix = xx + kx - PADK;
            if (ix < 0 || ix >= WW) continue;
            a += xp[iy * WW + ix] * wp[ky * 3 + kx];
        }
    }
    outT[((size_t)n * LL + l) * CDIM + c] = (_Float16)a;
}

// ---------------- WMMA GEMM (ping-pong pipelined, 64x(16*NT) tile/wave) ----
// C[m][n] = sum_k A[m][k] * Wt[n][k] + bias[n]
// A: [M, Kd] f16 row-major; Wt: [Nout, Kd] f16 row-major.
// Two named operand buffer sets alternate roles each k-step (no rotation
// copies -> no spill); loads of the next step overlap current WMMAs.
// Requires Kd % 64 == 0.
// mode 0: Cout[m*Nout + n] row-major; mode 1: NCHW transpose-on-store.
template <int NT>
__global__ void __launch_bounds__(256)
k_wmma_gemm(const _Float16* __restrict__ A, const _Float16* __restrict__ Wt,
            const float* __restrict__ bias, float* __restrict__ Cout,
            int Nout, int Kd, int nUnitsN, int totalUnits,
            int mode, int Lsz, int Couts) {
    const int lane = threadIdx.x & 31;
    const int unit = (int)(((size_t)blockIdx.x * blockDim.x + threadIdx.x) >> 5);
    if (unit >= totalUnits) return;   // wave-uniform: EXEC stays all-ones for WMMA

    const int mg = unit / nUnitsN;
    const int nu = unit % nUnitsN;
    const int m0 = mg * 64;
    const int n0 = nu * 16 * NT;

    const int laneM = lane & 15;
    const int klo   = (lane >> 4) * 8;      // K sub-block per lane half

    const _Float16* aP = A  + (size_t)(m0 + laneM) * Kd + klo;
    const _Float16* bP = Wt + (size_t)(n0 + laneM) * Kd + klo;

    v8f  acc[4][NT] = {};
    v16h a0[4], a1[4], b0[NT], b1[NT];

    // prologue: k-step 0 into set 0
    load_b_frags<NT>(b0, bP, 0, Kd);
    load_a_frags(a0, aP, 0, Kd);

    int k = 32;
#pragma unroll 1
    for (; k < Kd - 32; k += 64) {
        load_b_frags<NT>(b1, bP, k, Kd);          // fetch step k into set 1
        load_a_frags(a1, aP, k, Kd);
        mma_step<NT>(acc, a0, b0);                // compute step k-32
        load_b_frags<NT>(b0, bP, k + 32, Kd);     // fetch step k+32 into set 0
        load_a_frags(a0, aP, k + 32, Kd);
        mma_step<NT>(acc, a1, b1);                // compute step k
    }
    // tail: k == Kd-32; set 0 holds step Kd-64
    load_b_frags<NT>(b1, bP, k, Kd);
    load_a_frags(a1, aP, k, Kd);
    mma_step<NT>(acc, a0, b0);                    // step Kd-64
    mma_step<NT>(acc, a1, b1);                    // step Kd-32

    const int rsel = (lane >> 4) * 8;       // rows 0-7 vs 8-15 of each 16x16 tile
#pragma unroll
    for (int j = 0; j < NT; j++) {
        const int   colN = n0 + j * 16 + laneM;
        const float bv   = bias[colN];
#pragma unroll
        for (int i = 0; i < 4; i++) {
#pragma unroll
            for (int v = 0; v < 8; v++) {
                int   row = m0 + i * 16 + rsel + v;
                float val = acc[i][j][v] + bv;
                if (mode == 0) {
                    Cout[(size_t)row * Nout + colN] = val;
                } else {
                    int img = row / Lsz, l = row % Lsz;
                    Cout[((size_t)img * Couts + colN) * Lsz + l] = val;
                }
            }
        }
    }
}

// ---------------- deformable bilinear aggregation ----------------
// one wave per (n,l,g); lanes split the 64 group channels (2 each, float2)
__global__ void k_deform(const float* __restrict__ val, const float* __restrict__ om,
                         _Float16* __restrict__ accOut) {
    const int lane = threadIdx.x & 31;
    const int wid  = (int)(((size_t)blockIdx.x * blockDim.x + threadIdx.x) >> 5);
    if (wid >= NB * LL * GG) return;

    const int g = wid % GG;
    const int l = (wid / GG) % LL;
    const int n = wid / (GG * LL);

    const float yb = (float)(l >> 6) - (float)PADK;
    const float xb = (float)(l & 63) - (float)PADK;

    const float* omp   = om + ((size_t)n * LL + l) * PWO + g * (PP * 3);
    const int    cbase = g * CGRP + lane * 2;
    const float* vbase = val + (size_t)n * LL * CDIM + cbase;

    float2 acc; acc.x = 0.f; acc.y = 0.f;

    auto tap = [&](int iy, int ix, float w) {
        if (iy >= 0 && iy < HH && ix >= 0 && ix < WW) {
            const float2 v = *reinterpret_cast<const float2*>(vbase + (size_t)(iy * WW + ix) * CDIM);
            acc.x += w * v.x;
            acc.y += w * v.y;
        }
    };

#pragma unroll
    for (int p = 0; p < PP; p++) {
        const float offx = omp[p * 3 + 0];
        const float offy = omp[p * 3 + 1];
        const float mk   = omp[p * 3 + 2];
        const float posy = yb + (float)(p / 3) + offy;
        const float posx = xb + (float)(p % 3) + offx;
        const float y0f = floorf(posy), x0f = floorf(posx);
        const float fy = posy - y0f, fx = posx - x0f;
        const int iy0 = (int)y0f, ix0 = (int)x0f;
        tap(iy0,     ix0,     (1.f - fy) * (1.f - fx) * mk);
        tap(iy0,     ix0 + 1, (1.f - fy) * fx         * mk);
        tap(iy0 + 1, ix0,     fy         * (1.f - fx) * mk);
        tap(iy0 + 1, ix0 + 1, fy         * fx         * mk);
    }

    _Float16* op = accOut + ((size_t)n * LL + l) * CDIM + cbase;
    op[0] = (_Float16)acc.x;
    op[1] = (_Float16)acc.y;
}

// ---------------- launcher ----------------
extern "C" void kernel_launch(void* const* d_in, const int* in_sizes, int n_in,
                              void* d_out, int out_size, void* d_ws, size_t ws_size,
                              hipStream_t stream) {
    const float* x      = (const float*)d_in[0];
    const float* dw_w   = (const float*)d_in[1];
    const float* dw_b   = (const float*)d_in[2];
    const float* pw_w   = (const float*)d_in[3];
    const float* pw_b   = (const float*)d_in[4];
    const float* pin_w  = (const float*)d_in[5];
    const float* pin_b  = (const float*)d_in[6];
    const float* pout_w = (const float*)d_in[7];
    const float* pout_b = (const float*)d_in[8];
    float* out = (float*)d_out;

    // ---- workspace carve (all 256B aligned) ----
    char* ws = (char*)d_ws;
    size_t off = 0;
    auto carve = [&](size_t bytes) {
        void* p = ws + off;
        off += (bytes + 255) & ~(size_t)255;
        return p;
    };
    _Float16* xT     = (_Float16*)carve((size_t)MM * CDIM * 2);   // 8 MB
    _Float16* dwT    = (_Float16*)carve((size_t)MM * CDIM * 2);   // 8 MB
    _Float16* acc16  = (_Float16*)carve((size_t)MM * CDIM * 2);   // 8 MB
    _Float16* pinw16 = (_Float16*)carve((size_t)CDIM * CDIM * 2);
    _Float16* poutw16= (_Float16*)carve((size_t)CDIM * CDIM * 2);
    _Float16* pww16  = (_Float16*)carve((size_t)PWO * CDIM * 2);
    float*    valf   = (float*)carve((size_t)MM * CDIM * 4);      // 16 MB
    float*    omf    = (float*)carve((size_t)MM * PWO * 4);       // ~7 MB
    (void)ws_size;

    const int TB = 256;
    const size_t NCL = (size_t)NB * CDIM * LL;

    // stage activations + weights to f16
    k_xpose_f16<<<dim3((unsigned)((NCL + TB - 1) / TB)), dim3(TB), 0, stream>>>(x, xT);
    k_cvt_f16<<<dim3((CDIM * CDIM + TB - 1) / TB), dim3(TB), 0, stream>>>(pin_w, pinw16, CDIM * CDIM);
    k_cvt_f16<<<dim3((CDIM * CDIM + TB - 1) / TB), dim3(TB), 0, stream>>>(pout_w, poutw16, CDIM * CDIM);
    k_cvt_f16<<<dim3((PWO * CDIM + TB - 1) / TB), dim3(TB), 0, stream>>>(pw_w, pww16, PWO * CDIM);
    k_dwconv<<<dim3((unsigned)((NCL + TB - 1) / TB)), dim3(TB), 0, stream>>>(x, dw_w, dw_b, dwT);

    // GEMM 1: val = xT @ pin_w^T + pin_b   [16384 x 256], 64x32 tiles
    {
        int nUnitsN = CDIM / 32, units = (MM / 64) * nUnitsN;   // 8 * 256 = 2048
        int blocks = (units + 7) / 8;                            // 8 waves / block
        k_wmma_gemm<2><<<dim3(blocks), dim3(TB), 0, stream>>>(
            xT, pinw16, pin_b, valf, CDIM, CDIM, nUnitsN, units, 0, LL, CDIM);
    }
    // GEMM 2: om = dwT @ pw_w^T + pw_b     [16384 x 112], 64x16 tiles (7 n-tiles)
    {
        int nUnitsN = PWO / 16, units = (MM / 64) * nUnitsN;    // 7 * 256 = 1792
        int blocks = (units + 7) / 8;
        k_wmma_gemm<1><<<dim3(blocks), dim3(TB), 0, stream>>>(
            dwT, pww16, pw_b, omf, PWO, CDIM, nUnitsN, units, 0, LL, CDIM);
    }
    // deformable bilinear aggregation -> acc16 (f16 [N,L,C])
    {
        int waves = NB * LL * GG;                                // 65536
        int blocks = (waves + 7) / 8;
        k_deform<<<dim3(blocks), dim3(TB), 0, stream>>>(valf, omf, acc16);
    }
    // GEMM 3: out = acc @ pout_w^T + pout_b, stored NCHW  [16384 x 256]
    {
        int nUnitsN = CDIM / 32, units = (MM / 64) * nUnitsN;
        int blocks = (units + 7) / 8;
        k_wmma_gemm<2><<<dim3(blocks), dim3(TB), 0, stream>>>(
            acc16, poutw16, pout_b, out, CDIM, CDIM, nUnitsN, units, 1, LL, CDIM);
    }
    (void)in_sizes; (void)n_in; (void)out_size;
}
